// GINLayer_7413113553372
// MI455X (gfx1250) — matrix operand order repaired and checked by reading, not compiled, not run on hardware
//
#include <hip/hip_runtime.h>

#define D_FEAT   128
#define N_USER_C 100000
#define N_ITEM_C 100000
#define N_EDGE_C 600000
#define BN_EPS_C 1e-5f

typedef __attribute__((ext_vector_type(16))) _Float16 v16h;
typedef __attribute__((ext_vector_type(8)))  _Float16 v8h;
typedef __attribute__((ext_vector_type(8)))  float    v8f;

// Async global->LDS path (CDNA5): use if the toolchain exposes the builtins.
#if defined(__has_builtin)
#if __has_builtin(__builtin_amdgcn_global_load_async_to_lds_b128) && \
    __has_builtin(__builtin_amdgcn_s_wait_asynccnt)
#define USE_ASYNC_LDS 1
#endif
#endif
#ifndef USE_ASYNC_LDS
#define USE_ASYNC_LDS 0
#endif

#if USE_ASYNC_LDS
// Builtin signature (from compiler diagnostic): param0 is
// '__attribute__((vector_size(16))) int __device__ *' i.e. AS1 int32x4*.
typedef int async_b128_t __attribute__((vector_size(16)));
typedef __attribute__((address_space(1))) async_b128_t* async_gptr;
typedef __attribute__((address_space(3))) async_b128_t* async_lptr;
#endif

// ---------------------------------------------------------------------------
// float4 zero fill
// ---------------------------------------------------------------------------
__global__ void zero_f32_kernel(float* __restrict__ p, long n4) {
  long i = (long)blockIdx.x * blockDim.x + threadIdx.x;
  if (i < n4) reinterpret_cast<float4*>(p)[i] = make_float4(0.f, 0.f, 0.f, 0.f);
}

// ---------------------------------------------------------------------------
// x[dst[e],:] += h_src[src[e],:]   (one wave per edge, float4 gather + atomics)
// ---------------------------------------------------------------------------
__global__ void scatter_add_kernel(const float* __restrict__ hsrc,
                                   const int* __restrict__ src,
                                   const int* __restrict__ dst,
                                   float* __restrict__ x, int nedge) {
  int e = blockIdx.x * blockDim.y + threadIdx.y;
  if (e >= nedge) return;
  int s = src[e], d = dst[e];
  float4 v = reinterpret_cast<const float4*>(hsrc + (long)s * D_FEAT)[threadIdx.x];
  float* p = x + (long)d * D_FEAT + threadIdx.x * 4;
  atomicAdd(p + 0, v.x);
  atomicAdd(p + 1, v.y);
  atomicAdd(p + 2, v.z);
  atomicAdd(p + 3, v.w);
}

// ---------------------------------------------------------------------------
// Wt[n][k] (f16) = W[k][n] (f32) -- one-time transpose+convert per matmul
// ---------------------------------------------------------------------------
__global__ void wt_f16_kernel(const float* __restrict__ W,
                              _Float16* __restrict__ Wt) {
  int i = blockIdx.x * blockDim.x + threadIdx.x;   // 16384
  int k = i >> 7, nc = i & 127;
  Wt[nc * 128 + k] = (_Float16)W[i];
}

// ---------------------------------------------------------------------------
// Y[0:n,0:128] = A' @ W  with fused input transform and fused BN-stat epilogue.
//   MODE 0: A' = (1+eps)*A2 + A          (GIN combine fused into GEMM1)
//   MODE 1: A' = relu(g*(A-mu)*rs + b)   (BN1+ReLU fused into GEMM2)
// Epilogue: per-block column sum / sum-of-squares -> global f32 atomics.
// Block: 256 threads (8 waves), 128x128 tile; v_wmma_f32_16x16x32_f16.
// ---------------------------------------------------------------------------
template <int MODE>
__launch_bounds__(256)
__global__ void gemm128_wmma_kernel(const float* __restrict__ A,
                                    const float* __restrict__ A2,
                                    const float* __restrict__ eps,
                                    const _Float16* __restrict__ Wt,
                                    float* __restrict__ Y, int n,
                                    const float* __restrict__ mu,
                                    const float* __restrict__ rs,
                                    const float* __restrict__ g,
                                    const float* __restrict__ b,
                                    float* __restrict__ gsum,
                                    float* __restrict__ gsq) {
  __shared__ _Float16 lA[128 * 128];   // [row][k]
  __shared__ _Float16 lWt[128 * 128];  // [out-col][k]

  const int tid  = threadIdx.x;
  const long row0 = (long)blockIdx.x * 128;

  // ---- Stage Wt tile (raw f16, pre-transposed) into LDS -------------------
#if USE_ASYNC_LDS
  #pragma unroll
  for (int t = 0; t < 8; ++t) {
    int j = tid + t * 256;  // 2048 16-byte chunks
    __builtin_amdgcn_global_load_async_to_lds_b128(
        (async_gptr)(Wt + j * 8),
        (async_lptr)(&lWt[j * 8]), 0, 0);
  }
#else
  #pragma unroll
  for (int t = 0; t < 8; ++t) {
    int j = tid + t * 256;
    *reinterpret_cast<v8h*>(&lWt[j * 8]) =
        *reinterpret_cast<const v8h*>(Wt + j * 8);
  }
#endif

  // ---- Stage A tile with fused input transform, f32 -> f16 ----------------
  const float s1 = (MODE == 0) ? (1.0f + eps[0]) : 0.0f;
  for (int i = tid; i < 128 * 128; i += 256) {
    int r = i >> 7, k = i & 127;
    long gr = row0 + r;
    float v = 0.0f;
    if (gr < n) {
      if (MODE == 0) {
        v = s1 * A2[gr * D_FEAT + k] + A[gr * D_FEAT + k];
      } else {
        v = A[gr * D_FEAT + k];
        v = g[k] * (v - mu[k]) * rs[k] + b[k];
        v = v > 0.0f ? v : 0.0f;
      }
    }
    lA[i] = (_Float16)v;
  }
#if USE_ASYNC_LDS
  __builtin_amdgcn_s_wait_asynccnt(0);
#endif
  __syncthreads();

  // ---- WMMA main loop -----------------------------------------------------
  const int wave = tid >> 5;
  const int lane = tid & 31;
  const int hi   = lane >> 4;        // 0 or 1
  const int l15  = lane & 15;
  const int arow = wave * 16 + l15;
  const int kbA  = hi * 8;
  const int kbB  = hi * 8;

  v8f acc[8] = {};

  #pragma unroll
  for (int ks = 0; ks < 128; ks += 32) {
    union { v16h v; v8h h[2]; } af;
    af.h[0] = *reinterpret_cast<const v8h*>(&lA[arow * 128 + ks + kbA]);
    af.h[1] = *reinterpret_cast<const v8h*>(&lA[arow * 128 + ks + 16 + kbA]);
    #pragma unroll
    for (int nt = 0; nt < 8; ++nt) {
      const _Float16* bp = &lWt[(nt * 16 + l15) * 128 + ks + kbB];
      union { v16h v; v8h h[2]; } bf;
      bf.h[0] = *reinterpret_cast<const v8h*>(bp);
      bf.h[1] = *reinterpret_cast<const v8h*>(bp + 16);
      acc[nt] = __builtin_amdgcn_wmma_f32_16x16x32_f16(
          false, af.v, false, bf.v, (short)0, acc[nt], false, false);
    }
  }

  // ---- Fused BN-stat epilogue: per-block column sum / sumsq ---------------
  __syncthreads();                                  // all LDS reads done
  float* scol = reinterpret_cast<float*>(lA);       // reuse LDS: 256 floats
  scol[tid] = 0.0f;
  __syncthreads();

  const long rbase = row0 + wave * 16 + hi * 8;
  #pragma unroll
  for (int nt = 0; nt < 8; ++nt) {
    int gc = nt * 16 + l15;
    float s = 0.0f, q = 0.0f;
    #pragma unroll
    for (int r = 0; r < 8; ++r) {
      if (rbase + r < n) { float v = acc[nt][r]; s += v; q += v * v; }
    }
    atomicAdd(&scol[gc], s);          // ds_add_f32
    atomicAdd(&scol[128 + gc], q);
  }
  __syncthreads();
  if (tid < 128) {
    atomicAdd(&gsum[tid], scol[tid]);
    atomicAdd(&gsq[tid],  scol[128 + tid]);
  }

  // ---- Store Y ------------------------------------------------------------
  #pragma unroll
  for (int nt = 0; nt < 8; ++nt) {
    int gc = nt * 16 + l15;
    #pragma unroll
    for (int r = 0; r < 8; ++r) {
      long gr = rbase + r;
      if (gr < n) Y[gr * D_FEAT + gc] = acc[nt][r];
    }
  }
}

// ---------------------------------------------------------------------------
// mu/rs from accumulated sum/sumsq
// ---------------------------------------------------------------------------
__global__ void finalize_stats_kernel(const float* __restrict__ sum,
                                      const float* __restrict__ sumsq,
                                      float* __restrict__ mu,
                                      float* __restrict__ rs, int n) {
  int c = threadIdx.x;
  if (c < 128) {
    float m = sum[c] / (float)n;
    float v = sumsq[c] / (float)n - m * m;
    mu[c] = m;
    rs[c] = rsqrtf(v + BN_EPS_C);
  }
}

// ---------------------------------------------------------------------------
// out = [accumulate ? out : 0] + relu(g*(M-mu)*rs + b)
// ---------------------------------------------------------------------------
__global__ void bn_relu_out_kernel(const float* __restrict__ M, long total,
                                   const float* __restrict__ mu,
                                   const float* __restrict__ rs,
                                   const float* __restrict__ g,
                                   const float* __restrict__ b,
                                   float* __restrict__ out, int accumulate) {
  long i = (long)blockIdx.x * blockDim.x + threadIdx.x;
  if (i < total) {
    int c = (int)(i & 127);
    float v = g[c] * (M[i] - mu[c]) * rs[c] + b[c];
    v = v > 0.0f ? v : 0.0f;
    if (accumulate) out[i] += v;
    else out[i] = v;
  }
}

// ---------------------------------------------------------------------------
// Host orchestration (all on `stream`, graph-capture safe)
// ---------------------------------------------------------------------------
static void run_relation(const float* hsrc, const float* hdst,
                         const int* src, const int* dst, int ndst,
                         const float* W1, const float* W2,
                         const float* g1, const float* b1,
                         const float* g2, const float* b2,
                         const float* eps,
                         float* x, float* y, float* m, float* stats,
                         _Float16* wt1, _Float16* wt2,
                         float* out, int accumulate, hipStream_t stream) {
  float* ssum = stats;        // [128]
  float* ssq  = stats + 128;  // [128]
  float* smu  = stats + 256;  // [128]
  float* srs  = stats + 384;  // [128]

  // agg buffer: zero, then scatter h_src rows into it
  long n4 = (long)ndst * (D_FEAT / 4);
  zero_f32_kernel<<<(int)((n4 + 255) / 256), 256, 0, stream>>>(x, n4);
  dim3 sblk(32, 8);
  scatter_add_kernel<<<(N_EDGE_C + 7) / 8, sblk, 0, stream>>>(hsrc, src, dst, x,
                                                              N_EDGE_C);

  // pre-transpose/convert weights once per matmul
  wt_f16_kernel<<<64, 256, 0, stream>>>(W1, wt1);
  wt_f16_kernel<<<64, 256, 0, stream>>>(W2, wt2);

  int gblocks = (ndst + 127) / 128;

  // y = ((1+eps)*hdst + agg) @ W1, with BN1 stats accumulated in-epilogue
  zero_f32_kernel<<<1, 64, 0, stream>>>(stats, 64);   // zero sum/sumsq (256 f)
  gemm128_wmma_kernel<0><<<gblocks, 256, 0, stream>>>(
      x, hdst, eps, wt1, y, ndst, nullptr, nullptr, nullptr, nullptr, ssum, ssq);
  finalize_stats_kernel<<<1, 128, 0, stream>>>(ssum, ssq, smu, srs, ndst);

  // m = relu(BN1(y)) @ W2, with BN2 stats accumulated in-epilogue (m aliases x)
  zero_f32_kernel<<<1, 64, 0, stream>>>(stats, 64);
  gemm128_wmma_kernel<1><<<gblocks, 256, 0, stream>>>(
      y, nullptr, nullptr, wt2, m, ndst, smu, srs, g1, b1, ssum, ssq);
  finalize_stats_kernel<<<1, 128, 0, stream>>>(ssum, ssq, smu, srs, ndst);

  // out (+)= relu(BN2(m))
  long total = (long)ndst * D_FEAT;
  bn_relu_out_kernel<<<(int)((total + 255) / 256), 256, 0, stream>>>(
      m, total, smu, srs, g2, b2, out, accumulate);
}

extern "C" void kernel_launch(void* const* d_in, const int* in_sizes, int n_in,
                              void* d_out, int out_size, void* d_ws,
                              size_t ws_size, hipStream_t stream) {
  (void)in_sizes; (void)n_in; (void)out_size; (void)ws_size;

  const float* h_user      = (const float*)d_in[0];
  const float* h_item      = (const float*)d_in[1];
  const int*   src_follows = (const int*)d_in[2];
  const int*   dst_follows = (const int*)d_in[3];
  const int*   src_rates   = (const int*)d_in[4];
  const int*   dst_rates   = (const int*)d_in[5];
  const int*   src_ratedby = (const int*)d_in[6];
  const int*   dst_ratedby = (const int*)d_in[7];

  // params flattened in insertion order:
  //   relations: follows, rates, rated_by ; per-rel: W1, W2, g1, b1, g2, b2, eps
  const float* P[3][7];
  for (int r = 0; r < 3; ++r)
    for (int k = 0; k < 7; ++k)
      P[r][k] = (const float*)d_in[8 + r * 7 + k];

  const long MAXN = (N_USER_C > N_ITEM_C) ? N_USER_C : N_ITEM_C;
  float* x     = (float*)d_ws;         // [MAXN,128]  (reused as m)
  float* y     = x + MAXN * D_FEAT;    // [MAXN,128]
  float* m     = x;                    // alias: x is dead after GEMM1
  float* stats = y + MAXN * D_FEAT;    // 512 floats
  _Float16* wt1 = (_Float16*)(stats + 512);   // [128,128] f16
  _Float16* wt2 = wt1 + 128 * 128;            // [128,128] f16

  float* out_user = (float*)d_out;
  float* out_item = out_user + (long)N_USER_C * D_FEAT;

  // follows: user -> user (writes out_user)
  run_relation(h_user, h_user, src_follows, dst_follows, N_USER_C,
               P[0][0], P[0][1], P[0][2], P[0][3], P[0][4], P[0][5], P[0][6],
               x, y, m, stats, wt1, wt2, out_user, 0, stream);
  // rated_by: item -> user (accumulates into out_user)
  run_relation(h_item, h_user, src_ratedby, dst_ratedby, N_USER_C,
               P[2][0], P[2][1], P[2][2], P[2][3], P[2][4], P[2][5], P[2][6],
               x, y, m, stats, wt1, wt2, out_user, 1, stream);
  // rates: user -> item (writes out_item)
  run_relation(h_user, h_item, src_rates, dst_rates, N_ITEM_C,
               P[1][0], P[1][1], P[1][2], P[1][3], P[1][4], P[1][5], P[1][6],
               x, y, m, stats, wt1, wt2, out_item, 0, stream);
}
